// ErrorOptimizationPCN_64811056496642
// MI455X (gfx1250) — matrix operand order repaired and checked by reading, not compiled, not run on hardware
//
#include <hip/hip_runtime.h>

// ---------------------------------------------------------------------------
// Episodic-memory retrieval for MI455X (gfx1250).
//   scores = (4 * q/||q||) . keys[n]    (bf16 WMMA, f32 accumulate)
//   per-query top-16 over N=500k, softmax, weighted gather of values.
//
// Blocking: one block per key chunk, ALL B queries per block -> each key row
// is read from HBM exactly once (256 MB ~= 11 us at 23.3 TB/s), balanced
// against the bf16 matrix-core time for the 65.5 GFLOP score GEMM.
// Double-buffered LDS key tiles + score tiles -> one barrier per tile, with
// global->LDS staging overlapped by WMMA compute.
// ---------------------------------------------------------------------------

typedef __attribute__((ext_vector_type(16))) __bf16 v16bf;
typedef __attribute__((ext_vector_type(4)))  __bf16 v4bf;
typedef __attribute__((ext_vector_type(8)))  float  v8f;
typedef __attribute__((ext_vector_type(4)))  float  f4v;

#define KD      128
#define TOPK    16
#define NCHUNK  256
#define SCPAD   (TOPK + 1)              // score row stride: kill bank conflicts
#define NEG_INF (-3.0e38f)

__device__ __forceinline__ __bf16 bits_to_bf(unsigned short u) {
    return __builtin_bit_cast(__bf16, u);
}
__device__ __forceinline__ unsigned short f32_to_bf_bits(float f) {
    unsigned int u = __float_as_uint(f);
    u = (u + 0x7FFFu + ((u >> 16) & 1u)) >> 16;   // round-to-nearest-even
    return (unsigned short)u;
}

// ---------------------------------------------------------------------------
// Kernel 1: qn = (1/T-folded) normalized queries -> bf16 workspace
// ---------------------------------------------------------------------------
__global__ void __launch_bounds__(128) qnorm_kernel(const float* __restrict__ q,
                                                    unsigned short* __restrict__ qnb) {
    __shared__ float red[128];
    const int b = blockIdx.x;
    const int t = threadIdx.x;
    float x = q[(size_t)b * KD + t];
    red[t] = x * x;
    __syncthreads();
    for (int s = 64; s > 0; s >>= 1) {
        if (t < s) red[t] += red[t + s];
        __syncthreads();
    }
    float norm  = sqrtf(red[0]);
    float scale = 4.0f / fmaxf(norm, 1e-8f);   // 1/TEMPERATURE folded in
    qnb[(size_t)b * KD + t] = f32_to_bf_bits(x * scale);
}

// ---------------------------------------------------------------------------
// Kernel 2: one block per N-chunk, all B queries per block.
//   block = B threads (16 waves @ B=512); wave w owns q-tiles 2w, 2w+1 with
//   their bf16 A operands in registers. 16-key tiles staged once in LDS
//   (bf16, double-buffered) and reused by every wave. Thread t owns query
//   t's top-16. One barrier per tile; staging overlapped with WMMA.
// ---------------------------------------------------------------------------
__global__ void __launch_bounds__(512) score_topk_kernel(
    const unsigned short* __restrict__ qnb,
    const float*          __restrict__ keys,
    float*                __restrict__ cand_val,
    int*                  __restrict__ cand_idx,
    int N, int chunk_len)
{
    __shared__ __bf16 s_keys[2][16 * KD];        // 2 x 4 KB bf16 key tiles
    __shared__ float  s_sc[2][512][SCPAD];       // 2 x 512x16 scores (padded)

    const int tid  = threadIdx.x;
    const int w    = tid >> 5;
    const int lane = tid & 31;
    const int half = lane >> 4;
    const int lm   = lane & 15;
    const int qt0  = 2 * w;                      // first q-tile of this wave

    // ---- A operands: 2 q-tiles x 4 K-chunks, contiguous K order, same as B ---------
    // lane L: M = L&15 ; elems 0..15 -> K = half*16+e
    v16bf aq[2][4];
    #pragma unroll
    for (int qi = 0; qi < 2; ++qi) {
        const unsigned short* qrow = qnb + (size_t)((qt0 + qi) * 16 + lm) * KD;
        #pragma unroll
        for (int kc = 0; kc < 4; ++kc) {
            v16bf a;
            #pragma unroll
            for (int e = 0; e < 8; ++e) a[e]     = bits_to_bf(qrow[kc * 32 + half * 16 + e]);
            #pragma unroll
            for (int e = 0; e < 8; ++e) a[8 + e] = bits_to_bf(qrow[kc * 32 + half * 16 + 8 + e]);
            aq[qi][kc] = a;
        }
    }

    // ---- thread-private streaming top-16 for query `tid` -------------------
    float tv[TOPK];
    int   ti[TOPK];
    #pragma unroll
    for (int j = 0; j < TOPK; ++j) { tv[j] = NEG_INF; ti[j] = 0; }
    float tmin = NEG_INF;

    const int chunk_start = blockIdx.x * chunk_len;
    const int iters       = chunk_len >> 4;      // 16 keys per tile

    // staging indices: each of 512 threads loads 4 floats (one float4)
    const int st_key = (tid * 4) >> 7;           // key row in tile (0..15)
    const int st_k   = (tid * 4) & (KD - 1);     // K offset

    // ---- stage helper (lambda): tile `t` -> LDS buffer t&1 -----------------
    auto stage = [&](int t) {
        const int  row = chunk_start + t * 16 + st_key;
        const int  rc  = (row < N) ? row : (N - 1);
        const f4v  x   = *(const f4v*)(keys + (size_t)rc * KD + st_k);
        if (row + 16 < N)
            __builtin_prefetch(keys + (size_t)(row + 16) * KD + st_k, 0, 1);
        v4bf h;
        #pragma unroll
        for (int e = 0; e < 4; ++e) h[e] = (__bf16)x[e];
        *(v4bf*)(&s_keys[t & 1][0] + st_key * KD + st_k) = h;
    };

    // ---- top-16 scan helper: consume score tile `t` ------------------------
    auto scan = [&](int t) {
        const int nb0 = chunk_start + t * 16;
        #pragma unroll
        for (int col = 0; col < TOPK; ++col) {
            const float v = s_sc[t & 1][tid][col];
            if (v > tmin) {
                bool done = false;
                #pragma unroll
                for (int j2 = 0; j2 < TOPK; ++j2)
                    if (!done && tv[j2] == tmin) { tv[j2] = v; ti[j2] = nb0 + col; done = true; }
                float m = tv[0];
                #pragma unroll
                for (int j2 = 1; j2 < TOPK; ++j2) m = fminf(m, tv[j2]);
                tmin = m;
            }
        }
    };

    stage(0);
    __syncthreads();

    for (int it = 0; it < iters; ++it) {
        if (it + 1 < iters) stage(it + 1);       // overlap: next tile -> LDS

        // ---- B fragments (32x16 bf16: lane L: N=L&15, K=half*16+e) ---------
        const __bf16* kb = &s_keys[it & 1][0];
        v16bf bk[4];
        #pragma unroll
        for (int kc = 0; kc < 4; ++kc)
            bk[kc] = *(const v16bf*)(kb + lm * KD + kc * 32 + half * 16);

        const int  nb0   = chunk_start + it * 16;
        const bool valid = (nb0 + lm < N);

        #pragma unroll
        for (int qi = 0; qi < 2; ++qi) {
            v8f c = {0.f, 0.f, 0.f, 0.f, 0.f, 0.f, 0.f, 0.f};
            #pragma unroll
            for (int kc = 0; kc < 4; ++kc)
                c = __builtin_amdgcn_wmma_f32_16x16x32_bf16(false, aq[qi][kc],
                                                            false, bk[kc],
                                                            (short)0, c, false, false);
            // C layout: VGPR v, lanes0-15: M=v,N=lane ; lanes16-31: M=v+8
            #pragma unroll
            for (int v = 0; v < 8; ++v)
                s_sc[it & 1][(qt0 + qi) * 16 + v + half * 8][lm] = valid ? c[v] : NEG_INF;
        }

        if (it > 0) scan(it - 1);                // overlap: previous scores
        __syncthreads();                         // one barrier per tile
    }
    scan(iters - 1);

    // ---- thread t IS query t's chunk top-16: write candidates directly -----
    const size_t base = (size_t)tid * (NCHUNK * TOPK) + (size_t)blockIdx.x * TOPK;
    #pragma unroll
    for (int j = 0; j < TOPK; ++j) {
        cand_val[base + j] = tv[j];
        cand_idx[base + j] = ti[j];
    }
}

// ---------------------------------------------------------------------------
// Kernel 3: per-query merge of NCHUNK*16 candidates -> top16, softmax, gather
//   grid = B blocks, 256 threads
// ---------------------------------------------------------------------------
__global__ void __launch_bounds__(256) merge_out_kernel(
    const float* __restrict__ cand_val,
    const int*   __restrict__ cand_idx,
    const float* __restrict__ values,
    float*       __restrict__ out,
    int B, int VD)
{
    __shared__ float s_v[NCHUNK * TOPK];    // 4096 (16 KB)
    __shared__ int   s_i[NCHUNK * TOPK];    // 4096 (16 KB)
    __shared__ float r_v[256];
    __shared__ int   r_p[256];
    __shared__ float top_v[TOPK];
    __shared__ int   top_i[TOPK];
    __shared__ float tw[TOPK];

    const int qid  = blockIdx.x;
    const int t    = threadIdx.x;
    const int CAND = NCHUNK * TOPK;         // 4096

    for (int c = t; c < CAND; c += 256) {
        s_v[c] = cand_val[(size_t)qid * CAND + c];
        s_i[c] = cand_idx[(size_t)qid * CAND + c];
    }
    __syncthreads();

    for (int r = 0; r < TOPK; ++r) {
        float bm = NEG_INF; int bp = t * 16;
        #pragma unroll
        for (int j = 0; j < 16; ++j) {
            const int p = t * 16 + j;
            if (s_v[p] > bm) { bm = s_v[p]; bp = p; }
        }
        r_v[t] = bm; r_p[t] = bp;
        __syncthreads();
        if (t == 0) {
            float gm = NEG_INF; int gp = 0;
            for (int u = 0; u < 256; ++u)
                if (r_v[u] > gm) { gm = r_v[u]; gp = r_p[u]; }
            top_v[r] = gm;
            top_i[r] = s_i[gp];
            s_v[gp]  = NEG_INF;
        }
        __syncthreads();
    }

    if (t == 0) {
        const float m = top_v[0];           // sorted descending -> max
        float e[TOPK], sum = 0.f;
        #pragma unroll
        for (int k = 0; k < TOPK; ++k) { e[k] = __expf(top_v[k] - m); sum += e[k]; }
        const float inv = 1.0f / (sum + 1e-8f);
        #pragma unroll
        for (int k = 0; k < TOPK; ++k) tw[k] = e[k] * inv;
    }
    __syncthreads();

    if (t < VD) {
        float acc = 0.f;
        #pragma unroll
        for (int k = 0; k < TOPK; ++k)
            acc += tw[k] * values[(size_t)top_i[k] * VD + t];
        out[(size_t)qid * VD + t] = acc;
    }
    if (t < TOPK) {
        out[(size_t)B * VD + (size_t)qid * TOPK + t] = tw[t];
        out[(size_t)B * VD + (size_t)B * TOPK + (size_t)qid * TOPK + t] = (float)top_i[t];
    }
}

// ---------------------------------------------------------------------------
extern "C" void kernel_launch(void* const* d_in, const int* in_sizes, int n_in,
                              void* d_out, int out_size, void* d_ws, size_t ws_size,
                              hipStream_t stream) {
    (void)n_in; (void)out_size; (void)ws_size;
    const float* queries = (const float*)d_in[0];
    const float* keys    = (const float*)d_in[1];
    const float* values  = (const float*)d_in[2];

    const int B  = in_sizes[0] / KD;        // 512
    const int N  = in_sizes[1] / KD;        // 500000
    const int VD = in_sizes[2] / N;         // 64

    // chunk length: multiple of 16 covering N/NCHUNK
    const int chunk_len = (((N + NCHUNK - 1) / NCHUNK) + 15) & ~15;

    // workspace layout
    unsigned short* qnb = (unsigned short*)d_ws;
    size_t off = ((size_t)B * KD * sizeof(unsigned short) + 255) & ~(size_t)255;
    float* cand_val = (float*)((char*)d_ws + off);
    const size_t ncand = (size_t)B * NCHUNK * TOPK;
    int* cand_idx = (int*)((char*)d_ws + off + ncand * sizeof(float));

    qnorm_kernel<<<B, 128, 0, stream>>>(queries, qnb);

    score_topk_kernel<<<NCHUNK, B, 0, stream>>>(qnb, keys, cand_val, cand_idx,
                                                N, chunk_len);

    merge_out_kernel<<<B, 256, 0, stream>>>(cand_val, cand_idx, values,
                                            (float*)d_out, B, VD);
}
